// CustomMultiHead_74612171866275
// MI455X (gfx1250) — matrix operand order repaired and checked by reading, not compiled, not run on hardware
//
#include <hip/hip_runtime.h>
#include <stddef.h>

// ---------------------------------------------------------------------------
// out[M, 96] = x[M, 2048] @ W[96, 2048]^T + b      (M = 16384, fp32 in/out)
//
// MI455X plan:
//   * x (134 MB) streamed once from HBM  -> ~6 us floor at 23.3 TB/s
//   * bf16x3 split (a_hi*b_hi + a_lo*b_hi + a_hi*b_lo) on
//     V_WMMA_F32_16X16X32_BF16, f32 accumulate -> fp32-grade accuracy,
//     ~19 GFLOP of bf16 matrix work (well under the matrix roofline)
//   * W pre-split/pre-swizzled once (prepack kernel, d_ws), then staged
//     per-workgroup into LDS with global_load_async_to_lds_b128 +
//     s_wait_asynccnt (double buffered) -> B read from L2 once per BLOCK,
//     not once per WAVE (8x less L2 traffic), shared via LDS.
// ---------------------------------------------------------------------------

typedef __attribute__((ext_vector_type(16))) __bf16       v16bf;
typedef __attribute__((ext_vector_type(8)))  float        v8f;
typedef __attribute__((ext_vector_type(4)))  unsigned int v4u;

#define IN_F    2048
#define NHEADS  96
#define NT      6                      // 96/16 column tiles
#define KG      2                      // K-chunks (of 32) per LDS group
#define NGROUP  (IN_F / (32 * KG))     // 32 groups
#define TILE_STRIDE 1088               // 1KB B-tile + 64B swizzle slack (16B aligned)
#define GROUP_BYTES (KG * NT * 2 * TILE_STRIDE)   // 26112 B per group image
#define NB16        (GROUP_BYTES / 16)            // 1632 16B blocks per group
#define COPY_ITERS  ((NB16 + 255) / 256)          // 7

__device__ __forceinline__ unsigned short f2bf_rne(float f) {
    unsigned u = __float_as_uint(f);
    u += 0x7FFFu + ((u >> 16) & 1u);   // round-to-nearest-even to bf16
    return (unsigned short)(u >> 16);
}
__device__ __forceinline__ float bf2f(unsigned short h) {
    return __uint_as_float(((unsigned)h) << 16);
}
// 16B-granule swizzle: lanes 0-15 start at even granules, 16-31 at odd ->
// each 4-bank group serves exactly 2 lanes per ds_load_b128 phase (optimal).
__device__ __forceinline__ unsigned lane_swz(int lane) {
    return (unsigned)(lane * 32 + (lane >> 4) * 16);
}

union V16U { v16bf v; unsigned short u[16]; };
union BV   { v16bf v; v4u u2[2]; };

// ---------------------------------------------------------------------------
// Prepack W[96,2048] fp32 -> hi/lo bf16 "LDS image" in d_ws. Image is laid
// out exactly as it will sit in LDS (incl. swizzle + tile padding) so the
// async staging copy is a flat 16B-block memcpy.
// B-matrix (32x16 bf16) layout: lane n (0-15): col N=n, K=0..7,16..23 (8 VGPRs);
// lane 16+n: col N=n, K=8..15,24..31.
// One thread per 16B block: NGROUP*KG*NT*2*32*2 = 49152 threads.
// ---------------------------------------------------------------------------
__global__ void prepack_W_kernel(const float* __restrict__ W,
                                 unsigned char* __restrict__ img) {
    const int total = NGROUP * KG * NT * 2 * 32 * 2;
    int t = blockIdx.x * blockDim.x + threadIdx.x;
    if (t >= total) return;
    int q    = t & 1;            // which 16B half of the lane's 32B slot
    int lane = (t >> 1) & 31;
    int h    = (t >> 6) & 1;     // 0 = hi, 1 = lo
    int u    = t >> 7;           // ((g*KG + kc)*NT + nt)
    int nt   = u % NT;
    int v    = u / NT;           // g*KG + kc
    int kc   = v & (KG - 1);
    int g    = v >> 1;           // log2(KG) == 1

    int row = nt * 16 + (lane & 15);                      // head / column N
    int k0  = (g * KG + kc) * 32 + (lane >> 4) * 8 + q * 16;

    const float* src = W + (size_t)row * IN_F + k0;
    union { unsigned short s[8]; v4u v; } pk;
#pragma unroll
    for (int j = 0; j < 8; ++j) {
        float f = src[j];
        unsigned short hi = f2bf_rne(f);
        pk.s[j] = h ? f2bf_rne(f - bf2f(hi)) : hi;
    }
    size_t off = (size_t)g * GROUP_BYTES
               + (size_t)((kc * NT + nt) * 2 + h) * TILE_STRIDE
               + lane_swz(lane) + (unsigned)q * 16;
    *(v4u*)(img + off) = pk.v;
}

// ---------------------------------------------------------------------------
// Main GEMM: 256 threads = 8 waves; each wave owns a 16-row x 96-col strip.
// B double-buffered in LDS via async-to-LDS copies shared by all 8 waves.
// ---------------------------------------------------------------------------
__device__ __forceinline__ void copy_group_async(unsigned char* lds_dst,
                                                 const unsigned char* src,
                                                 int tid) {
#pragma unroll
    for (int i = 0; i < COPY_ITERS; ++i) {
        int j = i * 256 + tid;
        if (j < NB16) {
            unsigned dst = (unsigned)(size_t)(lds_dst + (size_t)j * 16);
            asm volatile("global_load_async_to_lds_b128 %0, %1, off"
                         :: "v"(dst), "v"(src + (size_t)j * 16)
                         : "memory");
        }
    }
}

__global__ __launch_bounds__(256) void
multihead_wmma_kernel(const float* __restrict__ x,
                      const unsigned char* __restrict__ Wimg,
                      const float* __restrict__ bias,
                      float* __restrict__ out,
                      int mtiles) {
    __shared__ __align__(16) unsigned char sB[2 * GROUP_BYTES];  // 52224 B

    const int tid  = threadIdx.x;
    const int lane = tid & 31;
    const int wave = tid >> 5;
    int mtile = blockIdx.x * 8 + wave;
    if (mtile >= mtiles) mtile = mtiles - 1;   // keep barriers wave-uniform

    const int lrow  = lane & 15;   // A row within tile / B column
    const int khalf = lane >> 4;

    v8f c[NT];
#pragma unroll
    for (int i = 0; i < NT; ++i)
        c[i] = (v8f){0.f, 0.f, 0.f, 0.f, 0.f, 0.f, 0.f, 0.f};

    // A 16-bit 16x32 layout (ISA): lanes 0-15 hold K {0..7}+{16..23},
    // lanes 16-31 hold K {8..15}+{24..31}  -> two 32B runs per lane.
    const float* xrow = x + (size_t)(mtile * 16 + lrow) * IN_F + khalf * 8;
    const unsigned swz = lane_swz(lane);

    // prologue: stage group 0
    copy_group_async(sB, Wimg, tid);

    for (int g = 0; g < NGROUP; ++g) {
        unsigned char* cur = sB + (unsigned)(g & 1) * GROUP_BYTES;
        unsigned char* nxt = sB + (unsigned)((g & 1) ^ 1) * GROUP_BYTES;

        asm volatile("s_wait_asynccnt 0" ::: "memory");
        __syncthreads();   // group g visible everywhere; g-1 fully consumed

        if (g + 1 < NGROUP)
            copy_group_async(nxt, Wimg + (size_t)(g + 1) * GROUP_BYTES, tid);

#pragma unroll
        for (int kc = 0; kc < KG; ++kc) {
            const float* xp = xrow + (size_t)(g * KG + kc) * 32;
            v8f a0 = *(const v8f*)(xp);        // K = khalf*8 .. +7
            v8f a1 = *(const v8f*)(xp + 16);   // K = 16 + khalf*8 .. +7
            V16U ahi, alo;
#pragma unroll
            for (int j = 0; j < 8; ++j) {
                float f0 = a0[j], f1 = a1[j];
                unsigned short h0 = f2bf_rne(f0);
                unsigned short h1 = f2bf_rne(f1);
                ahi.u[j]     = h0;  alo.u[j]     = f2bf_rne(f0 - bf2f(h0));
                ahi.u[8 + j] = h1;  alo.u[8 + j] = f2bf_rne(f1 - bf2f(h1));
            }
            const unsigned char* base =
                cur + (size_t)kc * (NT * 2 * TILE_STRIDE) + swz;
#pragma unroll
            for (int nt = 0; nt < NT; ++nt) {
                const unsigned char* p = base + (size_t)nt * (2 * TILE_STRIDE);
                BV bhi, blo;
                bhi.u2[0] = *(const v4u*)(p);
                bhi.u2[1] = *(const v4u*)(p + 16);
                blo.u2[0] = *(const v4u*)(p + TILE_STRIDE);
                blo.u2[1] = *(const v4u*)(p + TILE_STRIDE + 16);
                // D = A*B + C ; (neg_a, A, neg_b, B, c_mod, C, reuse_a, reuse_b)
                c[nt] = __builtin_amdgcn_wmma_f32_16x16x32_bf16(
                    false, ahi.v, false, bhi.v, (short)0, c[nt], false, false);
                c[nt] = __builtin_amdgcn_wmma_f32_16x16x32_bf16(
                    false, alo.v, false, bhi.v, (short)0, c[nt], false, false);
                c[nt] = __builtin_amdgcn_wmma_f32_16x16x32_bf16(
                    false, ahi.v, false, blo.v, (short)0, c[nt], false, false);
            }
        }
    }

    // epilogue: C layout -> lane l: N = l%16 ; VGPR r: M = r + 8*(l>=16)
    const int rbase = mtile * 16 + 8 * khalf;
#pragma unroll
    for (int nt = 0; nt < NT; ++nt) {
        const int col = nt * 16 + lrow;
        const float bv = bias[col];
#pragma unroll
        for (int r = 0; r < 8; ++r)
            out[(size_t)(rbase + r) * NHEADS + col] = c[nt][r] + bv;
    }
}

extern "C" void kernel_launch(void* const* d_in, const int* in_sizes, int n_in,
                              void* d_out, int out_size, void* d_ws, size_t ws_size,
                              hipStream_t stream) {
    const float* x = (const float*)d_in[0];   // [M, 2048]
    const float* W = (const float*)d_in[1];   // [96, 2048]
    const float* b = (const float*)d_in[2];   // [96]
    float* out = (float*)d_out;               // [M, 96]
    unsigned char* Wimg = (unsigned char*)d_ws;  // needs 32*26112 = 835584 B

    const int M = in_sizes[0] / IN_F;         // 16384
    const int mtiles = M / 16;                // 1024

    // 1) pre-split + pre-swizzle W into the LDS staging image (every call)
    const int pre_threads = NGROUP * KG * NT * 2 * 32 * 2;  // 49152
    prepack_W_kernel<<<(pre_threads + 255) / 256, 256, 0, stream>>>(W, Wimg);

    // 2) main streaming WMMA GEMM: 8 waves/block, LDS-shared B
    const int blocks = (mtiles + 7) / 8;      // 128
    multihead_wmma_kernel<<<blocks, 256, 0, stream>>>(x, Wimg, b, out, mtiles);
}
